// FrequencyEnhancementBranch_73710228734874
// MI455X (gfx1250) — matrix operand order repaired and checked by reading, not compiled, not run on hardware
//
#include <hip/hip_runtime.h>
#include <hip/hip_bf16.h>
#include <math.h>

typedef __attribute__((ext_vector_type(16))) __bf16 bf16x16;
typedef __attribute__((ext_vector_type(8)))  float  f32x8;
typedef __attribute__((address_space(3))) unsigned char lds_uchar;

#define CDIM 768
#define ODIM 768
#define HWN  64
#define GN_ELEMS (ODIM * HWN)      // 49152 elements per batch for GroupNorm(1)
#define CHUNK_K 384                // K staged into LDS in two chunks
#define LDSROW (CHUNK_K + 8)       // bf16 elems per LDS row: 784B stride, 16B aligned

// ---------------------------------------------------------------------------
// CDNA5 async global->LDS 128b copy (ASYNCcnt path, ISA §15.18.3 opcode 98).
// LDS destination passed as an address_space(3) pointer => one VGPR holding
// the raw LDS byte offset (VDST); global source is the 64-bit vaddr pair.
// ---------------------------------------------------------------------------
__device__ __forceinline__ void async_copy_b128(unsigned char* ldsp, const void* gp) {
  asm volatile("global_load_async_to_lds_b128 %0, %1, off"
               :
               : "v"((lds_uchar*)ldsp), "v"(gp)
               : "memory");
}

__device__ __forceinline__ void wait_asynccnt0() {
#if __has_builtin(__builtin_amdgcn_s_wait_asynccnt)
  __builtin_amdgcn_s_wait_asynccnt(0);
#else
  asm volatile("s_wait_asynccnt 0x0" ::: "memory");
#endif
}

// ---------------------------------------------------------------------------
// Kernel 0: convert proj_w f32 -> bf16 (stays resident in 192MB L2)
// ---------------------------------------------------------------------------
__global__ void cvt_bf16_kernel(const float* __restrict__ in,
                                __hip_bfloat16* __restrict__ out, int n) {
  int i = blockIdx.x * blockDim.x + threadIdx.x;
  if (i < n) out[i] = __float2bfloat16(in[i]);
}

// ---------------------------------------------------------------------------
// Kernel 1: per-(b,c) 8x8 rfft2 -> complex bandpass -> irfft2 (all fp32 VALU).
// ortho norms (1/8 fwd * 1/8 inv) folded into the weights as 1/64.
// Output written TRANSPOSED as [b][hw][c] in bf16 so that the GEMM B-matrix
// (K=c, N=hw) has its (K,K+1) 16-bit WMMA register pairs contiguous.
// ---------------------------------------------------------------------------
__global__ void __launch_bounds__(256)
fft_filter_kernel(const float* __restrict__ xin,   // (B, 64, C)
                  const float* __restrict__ wrin,  // (C, 8, 5)
                  const float* __restrict__ wiin,  // (C, 8, 5)
                  __hip_bfloat16* __restrict__ xoutT) { // (B, 64, C)
  const int c = blockIdx.x * blockDim.x + threadIdx.x;
  const int b = blockIdx.y;
  const size_t base = (size_t)b * HWN * CDIM + c;

  // x[b, c, h, w] = spatial_tokens[b, hw, c]  (coalesced across lanes in c)
  float x[64];
#pragma unroll
  for (int hw = 0; hw < 64; ++hw) x[hw] = xin[base + (size_t)hw * CDIM];

  float out[64];
#pragma unroll
  for (int i = 0; i < 64; ++i) out[i] = 0.0f;

  constexpr float R2 = 0.70710678118654752440f;
  const float COS8[8] = {1.f,  R2, 0.f, -R2, -1.f, -R2,  0.f,  R2};
  const float SIN8[8] = {0.f,  R2, 1.f,  R2,  0.f, -R2, -1.f, -R2};

#pragma unroll
  for (int f = 0; f < 5; ++f) {
    float wr[8], wi[8];
#pragma unroll
    for (int k = 0; k < 8; ++k) {
      wr[k] = wrin[(size_t)c * 40 + k * 5 + f] * (1.0f / 64.0f);
      wi[k] = wiin[(size_t)c * 40 + k * 5 + f] * (1.0f / 64.0f);
    }
    // forward row rfft (e^{-i 2pi w f / 8}), bin f, each row h
    float Rr[8], Ri[8];
#pragma unroll
    for (int h = 0; h < 8; ++h) {
      float sr = 0.f, si = 0.f;
#pragma unroll
      for (int w = 0; w < 8; ++w) {
        const int t = (w * f) & 7;
        sr += x[h * 8 + w] * COS8[t];
        si -= x[h * 8 + w] * SIN8[t];
      }
      Rr[h] = sr; Ri[h] = si;
    }
    // forward column DFT (e^{-i 2pi h k / 8}) + complex weight
    float Gr[8], Gi[8];
#pragma unroll
    for (int k = 0; k < 8; ++k) {
      float fr = 0.f, fi = 0.f;
#pragma unroll
      for (int h = 0; h < 8; ++h) {
        const int t = (h * k) & 7;
        fr += Rr[h] * COS8[t] + Ri[h] * SIN8[t];
        fi += Ri[h] * COS8[t] - Rr[h] * SIN8[t];
      }
      Gr[k] = fr * wr[k] - fi * wi[k];
      Gi[k] = fr * wi[k] + fi * wr[k];
    }
    // inverse column DFT (e^{+i 2pi h k / 8})
    float Zr[8], Zi[8];
#pragma unroll
    for (int h = 0; h < 8; ++h) {
      float zr = 0.f, zi = 0.f;
#pragma unroll
      for (int k = 0; k < 8; ++k) {
        const int t = (h * k) & 7;
        zr += Gr[k] * COS8[t] - Gi[k] * SIN8[t];
        zi += Gr[k] * SIN8[t] + Gi[k] * COS8[t];
      }
      Zr[h] = zr; Zi[h] = zi;
    }
    // inverse row with Hermitian reconstruction (bins 0 & 4 once, 1..3 doubled)
#pragma unroll
    for (int h = 0; h < 8; ++h) {
#pragma unroll
      for (int w = 0; w < 8; ++w) {
        if (f == 0)      out[h * 8 + w] += Zr[h];
        else if (f == 4) out[h * 8 + w] += (w & 1) ? -Zr[h] : Zr[h];
        else {
          const int t = (w * f) & 7;
          out[h * 8 + w] += 2.0f * (Zr[h] * COS8[t] - Zi[h] * SIN8[t]);
        }
      }
    }
  }

#pragma unroll
  for (int hw = 0; hw < 64; ++hw)
    xoutT[base + (size_t)hw * CDIM] = __float2bfloat16(out[hw]);
}

// ---------------------------------------------------------------------------
// Kernel 2: per-batch GEMM  y = proj_w (768x768) @ x_enh (768x64)  via
// v_wmma_f32_16x16x32_bf16, fused GroupNorm(1) + affine + exact GELU epilogue,
// transposed (B, 64, O) store. One workgroup (16 waves) per batch.
// B slab staged into LDS with global_load_async_to_lds_b128 (ASYNCcnt).
// ---------------------------------------------------------------------------
__global__ void __launch_bounds__(512)
gemm_gn_gelu_kernel(const __hip_bfloat16* __restrict__ Abf,   // (O, C) bf16
                    const __hip_bfloat16* __restrict__ xenhT, // (B, 64, C) bf16
                    const float* __restrict__ gnw,
                    const float* __restrict__ gnb,
                    float* __restrict__ out) {                // (B, 64, O)
  __shared__ __align__(16) unsigned char sB[HWN * LDSROW * 2]; // 50,176 B
  __shared__ float sGnW[ODIM];
  __shared__ float sGnB[ODIM];
  __shared__ float sSum, sSumSq;

  const int b    = blockIdx.x;
  const int tid  = threadIdx.x;
  const int lane = tid & 31;
  const int wv   = tid >> 5;   // wave 0..15 -> rows [wv*48, wv*48+48)
  const int half = lane >> 4;  // lane group within wave32
  const int l16  = lane & 15;

  if (tid == 0) { sSum = 0.f; sSumSq = 0.f; }
  for (int i = tid; i < ODIM; i += 512) { sGnW[i] = gnw[i]; sGnB[i] = gnb[i]; }

  f32x8 acc[3][4];
#pragma unroll
  for (int mt = 0; mt < 3; ++mt)
#pragma unroll
    for (int nt = 0; nt < 4; ++nt)
      acc[mt][nt] = (f32x8){0.f, 0.f, 0.f, 0.f, 0.f, 0.f, 0.f, 0.f};

  const size_t xbase = (size_t)b * HWN * CDIM;

  for (int kc = 0; kc < CDIM; kc += CHUNK_K) {
    __syncthreads(); // previous chunk fully consumed
    // stage 64 x CHUNK_K bf16 slab into padded LDS rows: async b128 copies,
    // no VGPR round-trip, tracked by ASYNCcnt.
    for (int q = tid; q < HWN * (CHUNK_K / 8); q += 512) {
      const int row = q / (CHUNK_K / 8);
      const int col = q % (CHUNK_K / 8);
      const unsigned char* gsrc = (const unsigned char*)xenhT +
                                  (xbase + (size_t)row * CDIM + kc) * 2 + col * 16;
      async_copy_b128(sB + row * (LDSROW * 2) + col * 16, gsrc);
    }
    wait_asynccnt0();   // this wave's async writes to LDS are visible
    __syncthreads();    // ... and every other wave's too

    for (int kk = 0; kk < CHUNK_K; kk += 32) {
      union { unsigned int u[8]; bf16x16 v; } afrag[3], bfrag[4];
      // A fragments (16x32 bf16): lane half 0 holds K 0..15,
      // half 1 holds K 16..31; each VGPR = (K, K+1) pair.
#pragma unroll
      for (int mt = 0; mt < 3; ++mt) {
        const int M = wv * 48 + mt * 16 + l16;
        const unsigned int* arow =
            (const unsigned int*)(Abf + (size_t)M * CDIM + (kc + kk));
#pragma unroll
        for (int v = 0; v < 8; ++v) {
          const int ko = 16 * half + 2 * v;
          afrag[mt].u[v] = arow[ko >> 1];
        }
      }
      // B fragments (32x16 bf16): lanes 0-15 K=0..15, lanes 16-31 K=16..31,
      // VGPR v = (K=16*half+2v, +1) pair at column N = l16 — contiguous in LDS.
#pragma unroll
      for (int nt = 0; nt < 4; ++nt) {
        const int N = nt * 16 + l16;
        const unsigned char* brow = sB + N * (LDSROW * 2) + kk * 2;
#pragma unroll
        for (int v = 0; v < 8; ++v) {
          const int ko = 16 * half + 2 * v;
          bfrag[nt].u[v] = *(const unsigned int*)(brow + ko * 2);
        }
      }
#pragma unroll
      for (int mt = 0; mt < 3; ++mt)
#pragma unroll
        for (int nt = 0; nt < 4; ++nt)
          acc[mt][nt] = __builtin_amdgcn_wmma_f32_16x16x32_bf16(
              false, afrag[mt].v, false, bfrag[nt].v,
              (short)0, acc[mt][nt], false, false);
    }
  }

  // ---- GroupNorm(1) stats straight from accumulators ----
  float s = 0.f, s2 = 0.f;
#pragma unroll
  for (int mt = 0; mt < 3; ++mt)
#pragma unroll
    for (int nt = 0; nt < 4; ++nt)
#pragma unroll
      for (int e = 0; e < 8; ++e) {
        const float v = acc[mt][nt][e];
        s += v; s2 += v * v;
      }
#pragma unroll
  for (int off = 16; off; off >>= 1) {
    s  += __shfl_xor(s,  off, 32);
    s2 += __shfl_xor(s2, off, 32);
  }
  if (lane == 0) { atomicAdd(&sSum, s); atomicAdd(&sSumSq, s2); } // ds_add_f32
  __syncthreads();

  const float mean = sSum * (1.0f / GN_ELEMS);
  const float var  = sSumSq * (1.0f / GN_ELEMS) - mean * mean;
  const float rstd = rsqrtf(var + 1e-5f);

  // ---- normalize + affine + exact GELU + transposed b128 stores ----
  // C/D layout: lanes 0-15 reg r -> M=r, lanes 16-31 -> M=r+8; N = l16.
#pragma unroll
  for (int mt = 0; mt < 3; ++mt) {
    const int obase = wv * 48 + mt * 16 + 8 * half;
#pragma unroll
    for (int nt = 0; nt < 4; ++nt) {
      const int hw = nt * 16 + l16;
      float r[8];
#pragma unroll
      for (int e = 0; e < 8; ++e) {
        float y = (acc[mt][nt][e] - mean) * rstd;
        y = y * sGnW[obase + e] + sGnB[obase + e];
        r[e] = 0.5f * y * (1.0f + erff(y * 0.70710678118654752440f));
      }
      float* op = out + ((size_t)b * HWN + hw) * ODIM + obase;
      *(float4*)(op)     = make_float4(r[0], r[1], r[2], r[3]);
      *(float4*)(op + 4) = make_float4(r[4], r[5], r[6], r[7]);
    }
  }
}

// ---------------------------------------------------------------------------
extern "C" void kernel_launch(void* const* d_in, const int* in_sizes, int n_in,
                              void* d_out, int out_size, void* d_ws, size_t ws_size,
                              hipStream_t stream) {
  const float* spatial = (const float*)d_in[0]; // (B, 64, 768)
  const float* wreal   = (const float*)d_in[1]; // (1, 768, 8, 5)
  const float* wimag   = (const float*)d_in[2];
  const float* projw   = (const float*)d_in[3]; // (768, 768)
  const float* gnw     = (const float*)d_in[4];
  const float* gnb     = (const float*)d_in[5];
  float* outp          = (float*)d_out;         // (B, 64, 768)

  const int B = in_sizes[0] / (HWN * CDIM);     // 1024

  __hip_bfloat16* Abf = (__hip_bfloat16*)d_ws;                       // 1.18 MB
  __hip_bfloat16* xT  = (__hip_bfloat16*)((char*)d_ws +
                                          (size_t)ODIM * CDIM * 2);  // 100.7 MB

  cvt_bf16_kernel<<<(ODIM * CDIM + 255) / 256, 256, 0, stream>>>(
      projw, Abf, ODIM * CDIM);
  fft_filter_kernel<<<dim3(CDIM / 256, B), 256, 0, stream>>>(
      spatial, wreal, wimag, xT);
  gemm_gn_gelu_kernel<<<B, 512, 0, stream>>>(Abf, xT, gnw, gnb, outp);
}
